// esn_73650099192365
// MI455X (gfx1250) — compile-verified, matrix operand order checked by reading, not run on hardware
//
#include <hip/hip_runtime.h>
#include <math.h>

// Problem dims (fixed by reference)
#define B_    64
#define T_    256
#define DIN_  512
#define DRES_ 2048
#define DOUT_ 512

typedef __attribute__((ext_vector_type(16))) __bf16 v16bf;
typedef __attribute__((ext_vector_type(8)))  float  v8f;

static constexpr size_t KT_RES = DRES_ / 32;  // 64 K-tiles over DRES
static constexpr size_t KT_IN  = DIN_  / 32;  // 16 K-tiles over DIN
static constexpr size_t MT     = B_    / 16;  // 4 M-tiles over batch
static constexpr size_t NT_RES = DRES_ / 16;  // 128 N-tiles over DRES
static constexpr size_t NT_OUT = DOUT_ / 16;  // 32 N-tiles over DOUT
static constexpr size_t FRAG   = 32 * 16;     // one 16x32 (or 32x16) bf16 tile = 512 elems

// Workspace layout (bf16 elements)
static constexpr size_t SZ_WRES  = NT_RES * KT_RES * FRAG;        // 4 Mi elems
static constexpr size_t SZ_WIN   = NT_RES * KT_IN  * FRAG;        // 1 Mi
static constexpr size_t SZ_WOUT  = NT_OUT * KT_RES * FRAG;        // 1 Mi
static constexpr size_t SZ_X     = (size_t)T_ * MT * KT_IN * FRAG;  // 8 Mi
static constexpr size_t SZ_STATE = (size_t)T_ * MT * KT_RES * FRAG; // 32 Mi
// total = 48,234,496 bf16 = ~92 MB of d_ws

// ISA 7.12.2: 16-bit A(16x32)/B(32x16) fragment: lane = (m|n)&15 + 16*khi,
// per-lane element j maps to K = (j<8 ? j : j+8) + 8*khi
__device__ __forceinline__ int kOfJ(int j, int khi) {
  return ((j < 8) ? j : (j + 8)) + khi * 8;
}

// ---------------- packing kernels (run once per launch) ----------------

// B-matrix fragments: dst[(nt*ktiles + kt)*512 + lane*16 + j] = W[(nt*16+n)*ldk + kt*32 + K]
__global__ __launch_bounds__(256) void pack_bfrag(const float* __restrict__ W,
                                                  __bf16* __restrict__ dst,
                                                  int ktiles, int ldk, size_t total) {
  for (size_t idx = (size_t)blockIdx.x * blockDim.x + threadIdx.x; idx < total;
       idx += (size_t)gridDim.x * blockDim.x) {
    int j    = (int)(idx & 15);
    int lane = (int)((idx >> 4) & 31);
    size_t rest = idx >> 9;
    int kt = (int)(rest % (size_t)ktiles);
    size_t nt = rest / (size_t)ktiles;
    int n   = lane & 15;
    int khi = lane >> 4;
    int k   = kt * 32 + kOfJ(j, khi);
    dst[idx] = (__bf16)W[(nt * 16 + (size_t)n) * (size_t)ldk + (size_t)k];
  }
}

// A-matrix fragments for x_in (B,T,DIN): dst[((t*MT+mt)*KT_IN + kt)*512 + lane*16 + j]
__global__ __launch_bounds__(256) void pack_afrag_x(const float* __restrict__ x,
                                                    __bf16* __restrict__ dst, size_t total) {
  for (size_t idx = (size_t)blockIdx.x * blockDim.x + threadIdx.x; idx < total;
       idx += (size_t)gridDim.x * blockDim.x) {
    int j    = (int)(idx & 15);
    int lane = (int)((idx >> 4) & 31);
    size_t rest = idx >> 9;
    int kt = (int)(rest & (KT_IN - 1));  rest >>= 4;
    int mt = (int)(rest & (MT - 1));     size_t t = rest >> 2;
    int m   = lane & 15;
    int khi = lane >> 4;
    int i   = kt * 32 + kOfJ(j, khi);
    int b   = mt * 16 + m;
    dst[idx] = (__bf16)x[((size_t)b * T_ + t) * DIN_ + (size_t)i];
  }
}

// A-matrix fragments for h_0 = x_res_init (B,DRES) into stateF[t=0]
__global__ __launch_bounds__(256) void pack_afrag_h0(const float* __restrict__ h0,
                                                     __bf16* __restrict__ dst, size_t total) {
  for (size_t idx = (size_t)blockIdx.x * blockDim.x + threadIdx.x; idx < total;
       idx += (size_t)gridDim.x * blockDim.x) {
    int j    = (int)(idx & 15);
    int lane = (int)((idx >> 4) & 31);
    size_t rest = idx >> 9;
    int kt = (int)(rest & (KT_RES - 1));
    int mt = (int)(rest >> 6);
    int m   = lane & 15;
    int khi = lane >> 4;
    int r   = kt * 32 + kOfJ(j, khi);
    int b   = mt * 16 + m;
    dst[idx] = (__bf16)h0[(size_t)b * DRES_ + (size_t)r];
  }
}

// ---------------- recurrent step: h_t = tanh(h_{t-1} Wres^T + x_{t-1} Win^T + b_in) ------
// grid (16, 4), block 256 (8 waves). Wave w handles ntile = bx*8+w, mtile = by.
__global__ __launch_bounds__(256) void esn_step(const __bf16* __restrict__ wresF,
                                                const __bf16* __restrict__ winF,
                                                const __bf16* __restrict__ xF,
                                                const float* __restrict__ b_in,
                                                __bf16* __restrict__ stateF, int t) {
  const int lane  = threadIdx.x & 31;
  const int wave  = threadIdx.x >> 5;
  const int ntile = blockIdx.x * 8 + wave;  // 0..127
  const int mtile = blockIdx.y;             // 0..3

  const __bf16* aRes = stateF + (((size_t)(t - 1) * MT + mtile) * KT_RES) * FRAG + (size_t)lane * 16;
  const __bf16* bRes = wresF  + ((size_t)ntile * KT_RES) * FRAG + (size_t)lane * 16;

  v8f acc = {};
#pragma unroll 4
  for (int kt = 0; kt < (int)KT_RES; ++kt) {
    v16bf a = *(const v16bf*)(aRes + (size_t)kt * FRAG);
    v16bf b = *(const v16bf*)(bRes + (size_t)kt * FRAG);
    acc = __builtin_amdgcn_wmma_f32_16x16x32_bf16(false, a, false, b, (short)0, acc, false, false);
  }

  // fused input projection: extra 16 K-tiles over DIN using x_{t-1}
  const __bf16* aIn = xF   + (((size_t)(t - 1) * MT + mtile) * KT_IN) * FRAG + (size_t)lane * 16;
  const __bf16* bIn = winF + ((size_t)ntile * KT_IN) * FRAG + (size_t)lane * 16;
#pragma unroll 4
  for (int kt = 0; kt < (int)KT_IN; ++kt) {
    v16bf a = *(const v16bf*)(aIn + (size_t)kt * FRAG);
    v16bf b = *(const v16bf*)(bIn + (size_t)kt * FRAG);
    acc = __builtin_amdgcn_wmma_f32_16x16x32_bf16(false, a, false, b, (short)0, acc, false, false);
  }

  const float bias = b_in[ntile * 16 + (lane & 15)];

  // scatter tanh result directly into next-step A-fragment layout.
  // C/D layout: element (m = v + 8*(lane>>4), n = lane&15) per VGPR v.
  // Output column r = ntile*16 + n lands in A-ktile = ntile>>1 at k_local = (ntile&1)*16 + n.
  const int k_local = (ntile & 1) * 16 + (lane & 15);
  const int khi2    = (k_local >> 3) & 1;
  const int j2      = (k_local & 7) + ((k_local >= 16) ? 8 : 0);
  __bf16* outBase = stateF + (((size_t)t * MT + mtile) * KT_RES + (size_t)(ntile >> 1)) * FRAG;
#pragma unroll
  for (int v = 0; v < 8; ++v) {
    float h = tanhf(acc[v] + bias);
    int m_local = v + (lane >> 4) * 8;          // 0..15
    outBase[(size_t)(m_local + khi2 * 16) * 16 + j2] = (__bf16)h;
  }
}

// ---------------- readout: y[b,t,:] = states[t,b,:] Wout^T + b_out ----------------
// grid (4, 4, 256), block 256 (8 waves). Wave w: otile = bx*8+w, mtile = by, t = bz.
__global__ __launch_bounds__(256) void esn_readout(const __bf16* __restrict__ stateF,
                                                   const __bf16* __restrict__ woutF,
                                                   const float* __restrict__ b_out,
                                                   float* __restrict__ out) {
  const int lane  = threadIdx.x & 31;
  const int wave  = threadIdx.x >> 5;
  const int otile = blockIdx.x * 8 + wave;  // 0..31
  const int mtile = blockIdx.y;             // 0..3
  const int t     = blockIdx.z;             // 0..255

  const __bf16* aS = stateF + (((size_t)t * MT + mtile) * KT_RES) * FRAG + (size_t)lane * 16;
  const __bf16* bW = woutF  + ((size_t)otile * KT_RES) * FRAG + (size_t)lane * 16;

  v8f acc = {};
#pragma unroll 4
  for (int kt = 0; kt < (int)KT_RES; ++kt) {
    v16bf a = *(const v16bf*)(aS + (size_t)kt * FRAG);
    v16bf b = *(const v16bf*)(bW + (size_t)kt * FRAG);
    acc = __builtin_amdgcn_wmma_f32_16x16x32_bf16(false, a, false, b, (short)0, acc, false, false);
  }

  const int n = otile * 16 + (lane & 15);
  const float bias = b_out[n];
#pragma unroll
  for (int v = 0; v < 8; ++v) {
    int m = mtile * 16 + v + (lane >> 4) * 8;  // batch index
    out[((size_t)m * T_ + t) * DOUT_ + n] = acc[v] + bias;
  }
}

// ---------------- host launcher ----------------
extern "C" void kernel_launch(void* const* d_in, const int* in_sizes, int n_in,
                              void* d_out, int out_size, void* d_ws, size_t ws_size,
                              hipStream_t stream) {
  (void)in_sizes; (void)n_in; (void)out_size; (void)ws_size;
  const float* x_res_init = (const float*)d_in[0];
  const float* x_in       = (const float*)d_in[1];
  const float* W_in       = (const float*)d_in[2];
  const float* b_in       = (const float*)d_in[3];
  const float* W_res      = (const float*)d_in[4];
  const float* W_out      = (const float*)d_in[5];
  const float* b_out      = (const float*)d_in[6];
  float* out = (float*)d_out;

  __bf16* ws     = (__bf16*)d_ws;
  __bf16* wresF  = ws;
  __bf16* winF   = wresF + SZ_WRES;
  __bf16* woutF  = winF  + SZ_WIN;
  __bf16* xF     = woutF + SZ_WOUT;
  __bf16* stateF = xF    + SZ_X;

  auto blocks = [](size_t total) { return (unsigned)((total + 255) / 256); };

  // one-time (per call) packing into WMMA fragment layouts
  pack_bfrag<<<blocks(SZ_WRES), 256, 0, stream>>>(W_res, wresF, (int)KT_RES, DRES_, SZ_WRES);
  pack_bfrag<<<blocks(SZ_WIN),  256, 0, stream>>>(W_in,  winF,  (int)KT_IN,  DIN_,  SZ_WIN);
  pack_bfrag<<<blocks(SZ_WOUT), 256, 0, stream>>>(W_out, woutF, (int)KT_RES, DRES_, SZ_WOUT);
  pack_afrag_x<<<blocks(SZ_X), 256, 0, stream>>>(x_in, xF, SZ_X);
  const size_t sz_h0 = MT * KT_RES * FRAG;
  pack_afrag_h0<<<blocks(sz_h0), 256, 0, stream>>>(x_res_init, stateF, sz_h0);

  // sequential recurrence: stream ordering provides inter-step dependency
  for (int t = 1; t < T_; ++t) {
    esn_step<<<dim3(NT_RES / 8, MT), 256, 0, stream>>>(wresF, winF, xF, b_in, stateF, t);
  }

  // fully parallel readout over all T states
  esn_readout<<<dim3(NT_OUT / 8, MT, T_), 256, 0, stream>>>(stateF, woutF, b_out, out);
}
